// QuantAttnBlock_11029476016410
// MI455X (gfx1250) — 16-bit and 8-bit matrix operand order repaired and checked by reading, compiled, not run on hardware
//
#include <hip/hip_runtime.h>
#include <hip/hip_bf16.h>
#include <math.h>
#include <stdint.h>

typedef __attribute__((ext_vector_type(16))) __bf16 v16bf;
typedef __attribute__((ext_vector_type(8)))  __bf16 v8bf;
typedef __attribute__((ext_vector_type(4)))  __bf16 v4bf;
typedef __attribute__((ext_vector_type(8)))  float  v8f;
typedef __attribute__((ext_vector_type(8)))  int    v8i;
typedef __attribute__((ext_vector_type(4)))  int    i4;
typedef __attribute__((ext_vector_type(2)))  int    i2;
typedef __attribute__((ext_vector_type(4)))  float  f4;
typedef __attribute__((ext_vector_type(4)))  unsigned int u32x4;
typedef __attribute__((ext_vector_type(8)))  int    i32x8;
typedef __attribute__((ext_vector_type(4)))  int    i32x4;

#define B_  4
#define C_  512
#define HW_ 4096
#define G_  32
#define CPG 16
#define EPS 1e-6f

#if defined(__has_builtin)
# if __has_builtin(__builtin_amdgcn_tensor_load_to_lds) && \
     __has_builtin(__builtin_amdgcn_s_wait_tensorcnt)
#  define HAVE_TDM 1
# endif
#endif

__device__ inline void tdm_issue_2d(unsigned lds_off, const void* gptr,
                                    unsigned tile_d0, unsigned tile_d1,
                                    unsigned tensor_d0, unsigned tensor_d1,
                                    unsigned long long stride_d0, unsigned lg2sz) {
#ifdef HAVE_TDM
  unsigned long long ga = (unsigned long long)(uintptr_t)gptr;
  u32x4 g0;
  g0[0] = 1u;
  g0[1] = lds_off;
  g0[2] = (unsigned)ga;
  g0[3] = (unsigned)((ga >> 32) & 0x01FFFFFFu) | (2u << 30);
  i32x8 g1;
  g1[0] = (int)(lg2sz << 16);
  g1[1] = (int)((tensor_d0 & 0xFFFFu) << 16);
  g1[2] = (int)((tensor_d0 >> 16) & 0xFFFFu) | (int)((tensor_d1 & 0xFFFFu) << 16);
  g1[3] = (int)((tensor_d1 >> 16) & 0xFFFFu) | (int)((tile_d0 & 0xFFFFu) << 16);
  g1[4] = (int)(tile_d1 & 0xFFFFu);
  g1[5] = (int)(stride_d0 & 0xFFFFFFFFull);
  g1[6] = (int)((stride_d0 >> 32) & 0xFFFFull);
  g1[7] = 0;
  i32x4 z4 = {0, 0, 0, 0};
# if __clang_major__ >= 23
  i32x8 z8 = {0, 0, 0, 0, 0, 0, 0, 0};
  __builtin_amdgcn_tensor_load_to_lds(g0, g1, z4, z4, z8, 0);
# else
  __builtin_amdgcn_tensor_load_to_lds(g0, g1, z4, z4, 0);
# endif
#else
  (void)lds_off; (void)gptr; (void)tile_d0; (void)tile_d1;
  (void)tensor_d0; (void)tensor_d1; (void)stride_d0; (void)lg2sz;
#endif
}

__device__ inline void coop_copy(char* lds, const char* g, int bytes) {
  for (int i = threadIdx.x * 16; i < bytes; i += 256 * 16)
    *(i4*)(lds + i) = *(const i4*)(g + i);
}
__device__ inline void coop_copy_2d(char* lds, const char* g, int rows,
                                    int rowbytes, size_t gstride) {
  int total = rows * rowbytes;
  for (int i = threadIdx.x * 16; i < total; i += 256 * 16) {
    int r = i / rowbytes, c = i - r * rowbytes;
    *(i4*)(lds + i) = *(const i4*)(g + (size_t)r * gstride + c);
  }
}

__device__ inline v16bf load_a_bf16_lds(const __bf16* p) {
  v8bf a0 = *(const v8bf*)p;
  v8bf a1 = *(const v8bf*)(p + 8);
  return __builtin_shufflevector(a0, a1, 0, 1, 2, 3, 4, 5, 6, 7,
                                 8, 9, 10, 11, 12, 13, 14, 15);
}
__device__ inline v8i load_b_iu8(const char* __restrict__ col, int k0, int half) {
  i4 lo = *(const i4*)(col + k0 + half * 16);
  i4 hi = *(const i4*)(col + k0 + 32 + half * 16);
  v8i b;
  b[0]=lo[0]; b[1]=lo[1]; b[2]=lo[2]; b[3]=lo[3];
  b[4]=hi[0]; b[5]=hi[1]; b[6]=hi[2]; b[7]=hi[3];
  return b;
}
__device__ inline v8i load_a_iu8(const char* __restrict__ row, int k0, int half) {
  v8i a;
#pragma unroll
  for (int j = 0; j < 8; j += 2) {
    i2 d = *(const i2*)(row + k0 + (j >> 2) * 32 + half * 16 + (j & 3) * 4);
    a[j] = d[0]; a[j + 1] = d[1];
  }
  return a;
}

__global__ __launch_bounds__(256) void wcvt_kernel(const float* __restrict__ s,
                                                   __bf16* __restrict__ d) {
  int i = (blockIdx.x * 256 + threadIdx.x) * 4;
  f4 v = *(const f4*)(s + i);
  v4bf o;
  o[0]=(__bf16)v[0]; o[1]=(__bf16)v[1]; o[2]=(__bf16)v[2]; o[3]=(__bf16)v[3];
  *(v4bf*)(d + i) = o;
}

__global__ __launch_bounds__(256) void gn_kernel(
    const float* __restrict__ x, const float* __restrict__ gamma,
    const float* __restrict__ beta, __bf16* __restrict__ hn_t) {
  int b = blockIdx.x / G_;
  int g = blockIdx.x % G_;
  int t = threadIdx.x;
  const int NEL = CPG * HW_;
  const float* xg = x + ((size_t)b * C_ + g * CPG) * HW_;

  float s = 0.f, ss = 0.f;
  for (int i = t; i < NEL; i += 256) { float v = xg[i]; s += v; ss += v * v; }
  __shared__ float rs[256], rss[256];
  rs[t] = s; rss[t] = ss;
  __syncthreads();
  for (int o = 128; o > 0; o >>= 1) {
    if (t < o) { rs[t] += rs[t + o]; rss[t] += rss[t + o]; }
    __syncthreads();
  }
  float mu   = rs[0] / (float)NEL;
  float var  = rss[0] / (float)NEL - mu * mu;
  float rsig = rsqrtf(var + EPS);

  for (int i = t; i < NEL; i += 256) {
    int c_l = i & (CPG - 1);
    int sp  = i >> 4;
    int c   = g * CPG + c_l;
    float v = xg[(size_t)c_l * HW_ + sp];
    float y = (v - mu) * rsig * gamma[c] + beta[c];
    hn_t[((size_t)b * HW_ + sp) * C_ + c] = (__bf16)y;
  }
}

__global__ __launch_bounds__(256) void qkv_kernel(
    const __bf16* __restrict__ hn_t,
    const __bf16* __restrict__ wqb, const __bf16* __restrict__ wkb,
    const __bf16* __restrict__ wvb,
    const float* __restrict__ bq, const float* __restrict__ bk,
    const float* __restrict__ bv,
    const float* __restrict__ dq, const float* __restrict__ zq,
    const float* __restrict__ dk, const float* __restrict__ zk,
    const float* __restrict__ dv, const float* __restrict__ zv,
    char* __restrict__ q8, char* __restrict__ k8, char* __restrict__ v8) {
  int wid  = threadIdx.x >> 5;
  int lane = threadIdx.x & 31;
  int half = lane >> 4, ln = lane & 15;
  int n_grp = blockIdx.x & 31;
  int m_t   = (blockIdx.x >> 5) & 31;
  int b     = (blockIdx.x >> 10) & 3;
  int mat   = blockIdx.x >> 12;
  int m0 = m_t * 16;
  int n0 = (n_grp * 8 + wid) * 16;

  const __bf16* Wb = mat == 0 ? wqb : (mat == 1 ? wkb : wvb);
  const float*  Bs = mat == 0 ? bq  : (mat == 1 ? bk  : bv);
  float delta = mat == 0 ? dq[0] : (mat == 1 ? dk[0] : dv[0]);
  int   izp   = (int)rintf(mat == 0 ? zq[0] : (mat == 1 ? zk[0] : zv[0]));
  float inv_delta = 1.f / delta;

  __shared__ __align__(16) __bf16 wlds[16 * C_];
  __shared__ __align__(16) unsigned char vtile[8][16][16];
#ifdef HAVE_TDM
  if (wid == 0) {
    tdm_issue_2d((unsigned)(uintptr_t)&wlds[0], Wb + (size_t)m0 * C_,
                 C_, 16, C_, C_, C_,  1);
    __builtin_amdgcn_s_wait_tensorcnt(0);
  }
#else
  coop_copy((char*)wlds, (const char*)(Wb + (size_t)m0 * C_), 16 * C_ * 2);
#endif
  __syncthreads();

  v8f acc = {};
  for (int k0 = 0; k0 < C_; k0 += 32) {
    v16bf a = load_a_bf16_lds(wlds + (size_t)ln * C_ + k0 + half * 16);
    const __bf16* hp = hn_t + ((size_t)b * HW_ + n0 + ln) * C_ + k0 + half * 16;
    v16bf bf = *(const v16bf*)hp;
    acc = __builtin_amdgcn_wmma_f32_16x16x32_bf16(false, a, false, bf,
                                                  (short)0, acc, false, false);
  }

  if (mat < 2) {
    char* out = mat == 0 ? q8 : k8;
    unsigned long long pack = 0;
#pragma unroll
    for (int i = 0; i < 8; ++i) {
      int M = m0 + i + 8 * half;
      float val = acc[i] + Bs[M];
      int xq = (int)rintf(val * inv_delta) + izp;
      xq = xq < 0 ? 0 : (xq > 255 ? 255 : xq);
      pack |= ((unsigned long long)(unsigned char)(char)(xq - izp)) << (8 * i);
    }
    *(unsigned long long*)(out + ((size_t)b * HW_ + n0 + ln) * C_ + m0 + 8 * half) = pack;
  } else {
#pragma unroll
    for (int i = 0; i < 8; ++i) {
      int M = m0 + i + 8 * half;
      float val = acc[i] + Bs[M];
      int xq = (int)rintf(val * inv_delta) + izp;
      xq = xq < 0 ? 0 : (xq > 255 ? 255 : xq);
      vtile[wid][i + 8 * half][ln] = (unsigned char)(char)(xq - izp);
    }
    __syncthreads();
    if (lane < 16) {
      i4 row = *(const i4*)&vtile[wid][lane][0];
      *(i4*)(v8 + ((size_t)b * C_ + m0 + lane) * HW_ + n0) = row;
    }
  }
}

__global__ __launch_bounds__(256) void attn_softmax_kernel(
    const char* __restrict__ q8, const char* __restrict__ k8,
    const float* __restrict__ dq, const float* __restrict__ dk,
    const float* __restrict__ dw, unsigned char* __restrict__ w8) {
  int wid  = threadIdx.x >> 5;
  int lane = threadIdx.x & 31;
  int half = lane >> 4, ln = lane & 15;
  int b     = blockIdx.x >> 5;
  int q_blk = blockIdx.x & 31;
  int q0 = q_blk * 128 + wid * 16;
  const int NT = HW_ / 16;
  const int TILEB = 16 * C_;

  float sf     = dq[0] * dk[0] * rsqrtf((float)C_);
  float inv_dw = 1.0f / dw[0];

  __shared__ __align__(16) char kbuf[2][16 * C_];
  __shared__ __align__(16) unsigned char ptile[8][16][16];

  v8i afr[8];
  const char* qrow = q8 + ((size_t)b * HW_ + q0 + ln) * C_;
#pragma unroll
  for (int st = 0; st < 8; ++st) afr[st] = load_a_iu8(qrow, st * 64, half);

  const char* kb = k8 + (size_t)b * HW_ * C_;

  float m[8], l[8];
#pragma unroll
  for (int i = 0; i < 8; ++i) { m[i] = -1e30f; l[i] = 0.f; }

#ifdef HAVE_TDM
  if (wid == 0)
    tdm_issue_2d((unsigned)(uintptr_t)&kbuf[0][0], kb, C_, 16, C_, HW_, C_, 0);
#endif
  for (int nt = 0; nt < NT; ++nt) {
    __syncthreads();
#ifdef HAVE_TDM
    if (wid == 0) {
      if (nt + 1 < NT) {
        tdm_issue_2d((unsigned)(uintptr_t)&kbuf[(nt + 1) & 1][0],
                     kb + (size_t)(nt + 1) * TILEB, C_, 16, C_, HW_, C_, 0);
        __builtin_amdgcn_s_wait_tensorcnt(1);
      } else {
        __builtin_amdgcn_s_wait_tensorcnt(0);
      }
    }
#else
    coop_copy(kbuf[nt & 1], kb + (size_t)nt * TILEB, TILEB);
#endif
    __syncthreads();
    const char* col = kbuf[nt & 1] + (size_t)ln * C_;
    v8i acc = {};
#pragma unroll
    for (int st = 0; st < 8; ++st)
      acc = __builtin_amdgcn_wmma_i32_16x16x64_iu8(true, afr[st], true,
                                                   load_b_iu8(col, st * 64, half),
                                                   acc, false, false);
#pragma unroll
    for (int i = 0; i < 8; ++i) {
      float s = (float)acc[i] * sf;
      float tm = s;
      tm = fmaxf(tm, __shfl_xor(tm, 1));
      tm = fmaxf(tm, __shfl_xor(tm, 2));
      tm = fmaxf(tm, __shfl_xor(tm, 4));
      tm = fmaxf(tm, __shfl_xor(tm, 8));
      float nm = fmaxf(m[i], tm);
      float e  = __expf(s - nm);
      e += __shfl_xor(e, 1); e += __shfl_xor(e, 2);
      e += __shfl_xor(e, 4); e += __shfl_xor(e, 8);
      l[i] = l[i] * __expf(m[i] - nm) + e;
      m[i] = nm;
    }
  }
  float invl[8];
#pragma unroll
  for (int i = 0; i < 8; ++i) invl[i] = l[i] > 0.f ? 1.f / l[i] : 0.f;

#ifdef HAVE_TDM
  if (wid == 0)
    tdm_issue_2d((unsigned)(uintptr_t)&kbuf[0][0], kb, C_, 16, C_, HW_, C_, 0);
#endif
  for (int nt = 0; nt < NT; ++nt) {
    __syncthreads();
#ifdef HAVE_TDM
    if (wid == 0) {
      if (nt + 1 < NT) {
        tdm_issue_2d((unsigned)(uintptr_t)&kbuf[(nt + 1) & 1][0],
                     kb + (size_t)(nt + 1) * TILEB, C_, 16, C_, HW_, C_, 0);
        __builtin_amdgcn_s_wait_tensorcnt(1);
      } else {
        __builtin_amdgcn_s_wait_tensorcnt(0);
      }
    }
#else
    coop_copy(kbuf[nt & 1], kb + (size_t)nt * TILEB, TILEB);
#endif
    __syncthreads();
    const char* col = kbuf[nt & 1] + (size_t)ln * C_;
    v8i acc = {};
#pragma unroll
    for (int st = 0; st < 8; ++st)
      acc = __builtin_amdgcn_wmma_i32_16x16x64_iu8(true, afr[st], true,
                                                   load_b_iu8(col, st * 64, half),
                                                   acc, false, false);
#pragma unroll
    for (int i = 0; i < 8; ++i) {
      float s = (float)acc[i] * sf;
      float p = __expf(s - m[i]) * invl[i];
      int u = (int)rintf(p * inv_dw);
      u = u < 0 ? 0 : (u > 255 ? 255 : u);
      ptile[wid][i + 8 * half][ln] = (unsigned char)u;
    }
    __syncthreads();
    if (lane < 16) {
      i4 row = *(const i4*)&ptile[wid][lane][0];
      *(i4*)(w8 + ((size_t)b * HW_ + q0 + lane) * HW_ + nt * 16) = row;
    }
  }
}

#define HV_NB 4
__global__ __launch_bounds__(256) void hv_kernel(
    const char* __restrict__ v8, const unsigned char* __restrict__ w8,
    const float* __restrict__ dv, const float* __restrict__ dw,
    __bf16* __restrict__ h_t) {
  int wid  = threadIdx.x >> 5;
  int lane = threadIdx.x & 31;
  int half = lane >> 4, ln = lane & 15;
  int n_grp = blockIdx.x & 7;
  int m_t   = (blockIdx.x >> 3) & 31;
  int b     = blockIdx.x >> 8;
  int m0 = m_t * 16;
  int n0 = (n_grp * 8 + wid) * (16 * HV_NB);
  float scale = dv[0] * dw[0];
  const int KH = HW_ / 2;

  __shared__ __align__(16) char vbuf[16 * (HW_ / 2)];

  const char* vpan = v8 + ((size_t)b * C_ + m0) * HW_;
  const char* wcol[HV_NB];
#pragma unroll
  for (int j = 0; j < HV_NB; ++j)
    wcol[j] = (const char*)w8 + ((size_t)b * HW_ + n0 + j * 16 + ln) * HW_;

  v8i acc[HV_NB] = {};
  for (int kh = 0; kh < 2; ++kh) {
    __syncthreads();
#ifdef HAVE_TDM
    if (wid == 0) {
      tdm_issue_2d((unsigned)(uintptr_t)&vbuf[0], vpan + kh * KH,
                   KH, 16, HW_, C_, HW_, 0);
      __builtin_amdgcn_s_wait_tensorcnt(0);
    }
#else
    coop_copy_2d(vbuf, vpan + kh * KH, 16, KH, HW_);
#endif
    __syncthreads();
    const char* vrow = vbuf + (size_t)ln * KH;
    for (int k0 = 0; k0 < KH; k0 += 64) {
      v8i a = load_a_iu8(vrow, k0, half);
#pragma unroll
      for (int j = 0; j < HV_NB; ++j)
        acc[j] = __builtin_amdgcn_wmma_i32_16x16x64_iu8(
            true, a, false, load_b_iu8(wcol[j] + kh * KH, k0, half),
            acc[j], false, false);
    }
  }
#pragma unroll
  for (int j = 0; j < HV_NB; ++j) {
    v8bf out;
#pragma unroll
    for (int i = 0; i < 8; ++i) out[i] = (__bf16)((float)acc[j][i] * scale);
    *(v8bf*)(h_t + ((size_t)b * HW_ + n0 + j * 16 + ln) * C_ + m0 + 8 * half) = out;
  }
}

__global__ __launch_bounds__(256) void proj_kernel(
    const __bf16* __restrict__ h_t, const __bf16* __restrict__ wpb,
    const float* __restrict__ bp, const float* __restrict__ x,
    float* __restrict__ out) {
  int wid  = threadIdx.x >> 5;
  int lane = threadIdx.x & 31;
  int half = lane >> 4, ln = lane & 15;
  int n_grp = blockIdx.x & 31;
  int m_t   = (blockIdx.x >> 5) & 31;
  int b     = blockIdx.x >> 10;
  int m0 = m_t * 16;
  int n0 = (n_grp * 8 + wid) * 16;

  __shared__ __align__(16) __bf16 wlds[16 * C_];
  __shared__ __align__(16) float otile[8][16][16];
#ifdef HAVE_TDM
  if (wid == 0) {
    tdm_issue_2d((unsigned)(uintptr_t)&wlds[0], wpb + (size_t)m0 * C_,
                 C_, 16, C_, C_, C_, 1);
    __builtin_amdgcn_s_wait_tensorcnt(0);
  }
#else
  coop_copy((char*)wlds, (const char*)(wpb + (size_t)m0 * C_), 16 * C_ * 2);
#endif
  __syncthreads();

  v8f acc = {};
  for (int k0 = 0; k0 < C_; k0 += 32) {
    v16bf a = load_a_bf16_lds(wlds + (size_t)ln * C_ + k0 + half * 16);
    const __bf16* hp = h_t + ((size_t)b * HW_ + n0 + ln) * C_ + k0 + half * 16;
    v16bf bf = *(const v16bf*)hp;
    acc = __builtin_amdgcn_wmma_f32_16x16x32_bf16(false, a, false, bf,
                                                  (short)0, acc, false, false);
  }
#pragma unroll
  for (int i = 0; i < 8; ++i)
    otile[wid][i + 8 * half][ln] = acc[i] + bp[m0 + i + 8 * half];
  __syncthreads();
  {
    int r = lane & 15, h = lane >> 4;
    f4 p0 = *(const f4*)&otile[wid][r][h * 8];
    f4 p1 = *(const f4*)&otile[wid][r][h * 8 + 4];
    size_t base = ((size_t)b * C_ + m0 + r) * HW_ + n0 + h * 8;
    f4 x0 = *(const f4*)(x + base);
    f4 x1 = *(const f4*)(x + base + 4);
    f4 o0, o1;
#pragma unroll
    for (int j = 0; j < 4; ++j) { o0[j] = x0[j] + p0[j]; o1[j] = x1[j] + p1[j]; }
    *(f4*)(out + base)     = o0;
    *(f4*)(out + base + 4) = o1;
  }
}

extern "C" void kernel_launch(void* const* d_in, const int* in_sizes, int n_in,
                              void* d_out, int out_size, void* d_ws, size_t ws_size,
                              hipStream_t stream) {
  (void)in_sizes; (void)n_in; (void)out_size;
  const float* x     = (const float*)d_in[0];
  const float* gamma = (const float*)d_in[1];
  const float* beta  = (const float*)d_in[2];
  const float* wq = (const float*)d_in[3];  const float* bq = (const float*)d_in[4];
  const float* wk = (const float*)d_in[5];  const float* bk = (const float*)d_in[6];
  const float* wv = (const float*)d_in[7];  const float* bv = (const float*)d_in[8];
  const float* wp = (const float*)d_in[9];  const float* bp = (const float*)d_in[10];
  const float* dq = (const float*)d_in[11]; const float* zq = (const float*)d_in[12];
  const float* dk = (const float*)d_in[13]; const float* zk = (const float*)d_in[14];
  const float* dv = (const float*)d_in[15]; const float* zv = (const float*)d_in[16];
  const float* dw = (const float*)d_in[17];

  char* wsb = (char*)d_ws;
  size_t off = 0;
  __bf16* wqb = (__bf16*)(wsb + off); off += (size_t)C_ * C_ * 2;
  __bf16* wkb = (__bf16*)(wsb + off); off += (size_t)C_ * C_ * 2;
  __bf16* wvb = (__bf16*)(wsb + off); off += (size_t)C_ * C_ * 2;
  __bf16* wpb = (__bf16*)(wsb + off); off += (size_t)C_ * C_ * 2;
  __bf16* hn_t = (__bf16*)(wsb + off); off += (size_t)B_ * HW_ * C_ * 2;
  char* q8 = wsb + off;                off += (size_t)B_ * HW_ * C_;
  char* k8 = wsb + off;                off += (size_t)B_ * HW_ * C_;
  char* v8 = wsb + off;                off += (size_t)B_ * C_ * HW_;
  unsigned char* w8 = (unsigned char*)(wsb + off); off += (size_t)B_ * HW_ * HW_;
  __bf16* h_t = (__bf16*)(wsb + off);  off += (size_t)B_ * HW_ * C_ * 2;
  if (off > ws_size) return;

  wcvt_kernel<<<256, 256, 0, stream>>>(wq, wqb);
  wcvt_kernel<<<256, 256, 0, stream>>>(wk, wkb);
  wcvt_kernel<<<256, 256, 0, stream>>>(wv, wvb);
  wcvt_kernel<<<256, 256, 0, stream>>>(wp, wpb);
  gn_kernel<<<B_ * G_, 256, 0, stream>>>(x, gamma, beta, hn_t);
  qkv_kernel<<<12288, 256, 0, stream>>>(hn_t, wqb, wkb, wvb, bq, bk, bv,
                                        dq, zq, dk, zk, dv, zv, q8, k8, v8);
  attn_softmax_kernel<<<B_ * 32, 256, 0, stream>>>(q8, k8, dq, dk, dw, w8);
  hv_kernel<<<1024, 256, 0, stream>>>(v8, w8, dv, dw, h_t);
  proj_kernel<<<4096, 256, 0, stream>>>(h_t, wpb, bp, x, (float*)d_out);
}
